// ParallelMagFaceLossAlt_77936476553556
// MI455X (gfx1250) — compile-verified, matrix operand order checked by reading, not run on hardware
//
#include <hip/hip_runtime.h>
#include <math.h>

// Problem constants (match reference)
#define B_  512
#define D_  512
#define C_  100000
#define U_M 0.8f
#define L_M 0.45f
#define U_A 110.0f
#define L_A 10.0f
#define LAMBDA_G 35.0f
#define SCALE_ 64.0f

#define NCOL_BLK 128                 // classes per block (16 per wave * 8 waves)
#define NBLK ((C_ + NCOL_BLK - 1) / NCOL_BLK)   // 782
#define APAD 520                     // padded LDS row stride in halfs (16B aligned, conflict-free)
#define LDSA_BYTES (32 * APAD * 2)   // one 32-row x_hat stage buffer = 33280 B
#define SMEM_BYTES (2 * LDSA_BYTES + 8 * B_ * 4)   // 2 stage buffers + per-wave sumexp = 82944 B

typedef _Float16 v16h __attribute__((ext_vector_type(16)));
typedef _Float16 v8h  __attribute__((ext_vector_type(8)));
typedef _Float16 v4h  __attribute__((ext_vector_type(4)));
typedef float    v8f  __attribute__((ext_vector_type(8)));

union FragU { v16h v; v8h h[2]; };

__device__ __forceinline__ void cvt4(v16h& v, int i, float4 f) {
    v[i+0] = (_Float16)f.x; v[i+1] = (_Float16)f.y;
    v[i+2] = (_Float16)f.z; v[i+3] = (_Float16)f.w;
}
__device__ __forceinline__ float sq4(float4 f) {
    return f.x*f.x + f.y*f.y + f.z*f.z + f.w*f.w;
}

// Async copy: 16B per lane, global -> LDS, tracked by ASYNCcnt (CDNA5 path).
__device__ __forceinline__ void async_b128(const _Float16* gsrc, _Float16* ldst) {
    unsigned lds_off = (unsigned)(uintptr_t)ldst;                 // LDS aperture: addr[31:0] = offset
    unsigned long long gaddr = (unsigned long long)(uintptr_t)gsrc;
    asm volatile("global_load_async_to_lds_b128 %0, %1, off"
                 :: "v"(lds_off), "v"(gaddr) : "memory");
}
__device__ __forceinline__ void wait_async0() {
    asm volatile("s_wait_asynccnt 0x0" ::: "memory");
}

// ---------------------------------------------------------------------------
// Kernel 1: normalize x rows -> fp16 x_hat; per-row margin; zero tgt logits.
// ---------------------------------------------------------------------------
__global__ __launch_bounds__(128)
void prep_kernel(const float* __restrict__ x, const float* __restrict__ xnorm,
                 _Float16* __restrict__ xhat, float* __restrict__ margin,
                 float* __restrict__ row_tgt) {
    int row = blockIdx.x;
    int t   = threadIdx.x;
    float4 v = ((const float4*)(x + row * D_))[t];
    float ss = sq4(v);
    #pragma unroll
    for (int m = 1; m < 32; m <<= 1) ss += __shfl_xor(ss, m, 32);
    __shared__ float red[4];
    if ((t & 31) == 0) red[t >> 5] = ss;
    __syncthreads();
    float tot = red[0] + red[1] + red[2] + red[3];
    float inv = rsqrtf(fmaxf(tot, 1e-24f));   // 1/max(||x||, 1e-12)
    v4h hv;
    hv[0] = (_Float16)(v.x * inv); hv[1] = (_Float16)(v.y * inv);
    hv[2] = (_Float16)(v.z * inv); hv[3] = (_Float16)(v.w * inv);
    *(v4h*)(xhat + row * D_ + t * 4) = hv;
    if (t == 0) {
        float xn = xnorm[row];
        margin[row]  = (U_M - L_M) / (U_A - L_A) * (xn - L_A) + L_M;
        row_tgt[row] = 0.0f;
    }
}

// ---------------------------------------------------------------------------
// Kernel 2: fused GEMM + MagFace epilogue + partial softmax sums.
// Block owns 128 classes x all 512 rows (W read exactly once from HBM).
// x_hat staged via double-buffered GLOBAL_LOAD_ASYNC_TO_LDS_B128.
// Two independent accumulator chains per wave hide LDS latency behind WMMA.
// ---------------------------------------------------------------------------
__global__ __launch_bounds__(256)
void magface_gemm_kernel(const float* __restrict__ W,
                         const _Float16* __restrict__ xhat,
                         const int* __restrict__ labels,
                         const float* __restrict__ margin,
                         float* __restrict__ row_tgt,
                         float* __restrict__ partial) {
    extern __shared__ char smem[];
    _Float16* ldsA0    = (_Float16*)(smem);
    _Float16* ldsA1    = (_Float16*)(smem + LDSA_BYTES);
    float*    sumexp_w = (float*)   (smem + 2 * LDSA_BYTES);   // [8][512]

    const int t    = threadIdx.x;
    const int wave = t >> 5;
    const int lane = t & 31;
    const int blk  = blockIdx.x;
    const int n0   = blk * NCOL_BLK + wave * 16;

    // zero per-wave exp-sum accumulators
    #pragma unroll
    for (int i = 0; i < 16; ++i) sumexp_w[t + i * 256] = 0.0f;

    // ---- Resident B fragments (16 ktiles of K=32) + per-class sumsq ----
    const int  ncol = n0 + (lane & 15);
    const int  nn   = (ncol < C_) ? ncol : (C_ - 1);
    const bool cvalid = (ncol < C_);
    const float* wrow = W + (size_t)nn * D_;
    const int khalf = (lane >> 4) * 8;

    FragU bfrag[16];
    float ss = 0.0f;
    #pragma unroll
    for (int kt = 0; kt < 16; ++kt) {
        int kb = kt * 32 + khalf;
        float4 f0 = *(const float4*)(wrow + kb);
        float4 f1 = *(const float4*)(wrow + kb + 4);
        float4 f2 = *(const float4*)(wrow + kb + 16);
        float4 f3 = *(const float4*)(wrow + kb + 20);
        ss += sq4(f0) + sq4(f1) + sq4(f2) + sq4(f3);
        v16h bv;
        cvt4(bv, 0, f0); cvt4(bv, 4, f1); cvt4(bv, 8, f2); cvt4(bv, 12, f3);
        bfrag[kt].v = bv;
    }
    ss += __shfl_xor(ss, 16, 32);               // lane-pair shares the class
    const float invn = rsqrtf(fmaxf(ss, 1e-24f));

    // ---- async stage of one 32-row x_hat group (1024 x 16B chunks) ----
    auto stage = [&](int mg, _Float16* dst) {
        #pragma unroll
        for (int i = 0; i < 4; ++i) {
            int c = t + i * 256;
            int r = c >> 5, kc = (c & 31) * 8;
            async_b128(xhat + (size_t)(mg * 32 + r) * D_ + kc, &dst[r * APAD + kc]);
        }
    };

    // ---- fused MagFace epilogue for one 16x16 accumulator tile ----
    auto epilogue = [&](v8f acc, int rowbase) {
        #pragma unroll
        for (int v = 0; v < 8; ++v) {
            int row = rowbase + v;
            float cosv = fminf(fmaxf(acc[v] * invn, -1.0f), 1.0f);
            float logit = cosv * SCALE_;
            if (cvalid && (labels[row] == ncol)) {
                float th = acosf(cosv) + margin[row];
                logit = __cosf(th) * SCALE_;
                row_tgt[row] = logit;           // unique writer across the grid
            }
            float e = cvalid ? __expf(logit - SCALE_) : 0.0f;
            e += __shfl_xor(e, 1, 32);
            e += __shfl_xor(e, 2, 32);
            e += __shfl_xor(e, 4, 32);
            e += __shfl_xor(e, 8, 32);
            if ((lane & 15) == 0)               // distinct rows per half-wave: race-free
                sumexp_w[wave * B_ + row] += e;
        }
    };

    stage(0, ldsA0);                            // prologue

    #pragma unroll 1
    for (int mg = 0; mg < 16; ++mg) {
        wait_async0();                          // my stage of mg has landed
        __syncthreads();                        // everyone's stage landed; prev compute done
        _Float16* cur = (mg & 1) ? ldsA1 : ldsA0;
        if (mg < 15) stage(mg + 1, (mg & 1) ? ldsA0 : ldsA1);   // overlap next copy w/ compute

        // two M-tiles as independent accumulator chains sharing bfrag[kt]
        v8f acc0 = {0.f, 0.f, 0.f, 0.f, 0.f, 0.f, 0.f, 0.f};
        v8f acc1 = {0.f, 0.f, 0.f, 0.f, 0.f, 0.f, 0.f, 0.f};
        const _Float16* arow0 = &cur[(lane & 15) * APAD + khalf];
        const _Float16* arow1 = arow0 + 16 * APAD;

        #pragma unroll
        for (int kt = 0; kt < 16; ++kt) {
            const _Float16* p0 = arow0 + kt * 32;
            const _Float16* p1 = arow1 + kt * 32;
            FragU a0, a1;
            a0.h[0] = *(const v8h*)(p0);
            a0.h[1] = *(const v8h*)(p0 + 16);
            a1.h[0] = *(const v8h*)(p1);
            a1.h[1] = *(const v8h*)(p1 + 16);
            acc0 = __builtin_amdgcn_wmma_f32_16x16x32_f16(
                    false, a0.v, false, bfrag[kt].v, (short)0, acc0, false, false);
            acc1 = __builtin_amdgcn_wmma_f32_16x16x32_f16(
                    false, a1.v, false, bfrag[kt].v, (short)0, acc1, false, false);
        }

        int rb = mg * 32 + ((lane >> 4) << 3);
        epilogue(acc0, rb);
        epilogue(acc1, rb + 16);
    }

    __syncthreads();
    // deterministic per-block partial: fixed-order sum over the 8 waves
    #pragma unroll
    for (int i = 0; i < 2; ++i) {
        int r = t + i * 256;
        float s = 0.0f;
        #pragma unroll
        for (int w = 0; w < 8; ++w) s += sumexp_w[w * B_ + r];
        partial[(size_t)blk * B_ + r] = s;
    }
}

// ---------------------------------------------------------------------------
// Kernel 3: reduce partials (fixed order), CE + lambda * G-loss -> scalar.
// ---------------------------------------------------------------------------
__global__ __launch_bounds__(256)
void finalize_kernel(const float* __restrict__ partial,
                     const float* __restrict__ row_tgt,
                     const float* __restrict__ xnorm,
                     float* __restrict__ out, int nblk) {
    int t = threadIdx.x;
    float nll_s = 0.0f, g_s = 0.0f;
    for (int r = t; r < B_; r += 256) {
        float s = 0.0f;
        for (int b = 0; b < nblk; ++b) s += partial[(size_t)b * B_ + r];
        nll_s += logf(s) + SCALE_ - row_tgt[r];     // lse(shift=64) - tgt
        float xn = xnorm[r];
        g_s += xn * (1.0f / (U_A * U_A)) + 1.0f / xn;
    }
    #pragma unroll
    for (int m = 1; m < 32; m <<= 1) {
        nll_s += __shfl_xor(nll_s, m, 32);
        g_s   += __shfl_xor(g_s,   m, 32);
    }
    __shared__ float rn[8], rg[8];
    if ((t & 31) == 0) { rn[t >> 5] = nll_s; rg[t >> 5] = g_s; }
    __syncthreads();
    if (t == 0) {
        float n = 0.0f, g = 0.0f;
        #pragma unroll
        for (int i = 0; i < 8; ++i) { n += rn[i]; g += rg[i]; }
        out[0] = n / (float)B_ + LAMBDA_G * (g / (float)B_);
    }
}

// ---------------------------------------------------------------------------
extern "C" void kernel_launch(void* const* d_in, const int* in_sizes, int n_in,
                              void* d_out, int out_size, void* d_ws, size_t ws_size,
                              hipStream_t stream) {
    const float* x      = (const float*)d_in[0];     // [512,512] f32
    const float* xnorm  = (const float*)d_in[1];     // [512]     f32
    const int*   labels = (const int*)  d_in[2];     // [512]     int
    const float* W      = (const float*)d_in[3];     // [100000,512] f32
    float* out = (float*)d_out;

    // workspace layout (~2.13 MB total)
    char* ws = (char*)d_ws;
    _Float16* xhat   = (_Float16*)ws;                         // 512*512*2 = 524288
    float*    margin = (float*)(ws + 524288);                 // 2048
    float*    tgt    = (float*)(ws + 526336);                 // 2048
    float*    part   = (float*)(ws + 528384);                 // NBLK*512*4

    prep_kernel<<<B_, 128, 0, stream>>>(x, xnorm, xhat, margin, tgt);
    magface_gemm_kernel<<<NBLK, 256, SMEM_BYTES, stream>>>(W, xhat, labels, margin, tgt, part);
    finalize_kernel<<<1, 256, 0, stream>>>(part, tgt, xnorm, out, NBLK);
}
